// Block_17128329576499
// MI455X (gfx1250) — matrix operand order repaired and checked by reading, not compiled, not run on hardware
//
#include <hip/hip_runtime.h>

// Problem constants
#define Bdim 8
#define Tdim 1024
#define Cdim 1024
#define Hdim 16
#define Ddim 64
#define Fdim 4096
#define Mrows (Bdim * Tdim)

typedef __attribute__((ext_vector_type(16))) __bf16 v16bf;
typedef __attribute__((ext_vector_type(8)))  __bf16 v8bf;
typedef __attribute__((ext_vector_type(8)))  float  v8f;
typedef __attribute__((ext_vector_type(4)))  int    v4i;

union Frag16 { v16bf v; v8bf h[2]; };

__device__ __forceinline__ v8f wmma_bf16(Frag16 a, Frag16 b, v8f c) {
  // D = A(16x32 bf16) * B(32x16 bf16) + C(16x16 f32)
  return __builtin_amdgcn_wmma_f32_16x16x32_bf16(
      /*neg_a=*/false, a.v, /*neg_b=*/false, b.v,
      /*c_mod=*/(short)0, c, /*reuse_a=*/false, /*reuse_b=*/false);
}

// ---- Async global->LDS staging (gfx1250 GLOBAL_LOAD_ASYNC_TO_LDS_B128) ----
#define AS1 __attribute__((address_space(1)))
#define AS3 __attribute__((address_space(3)))

#if defined(__has_builtin)
#if __has_builtin(__builtin_amdgcn_global_load_async_to_lds_b128) && \
    __has_builtin(__builtin_amdgcn_s_wait_asynccnt)
#define USE_ASYNC_LDS 1
#endif
#endif

__device__ __forceinline__ void stage16(const __bf16* g, __bf16* l) {
#ifdef USE_ASYNC_LDS
  // per-lane 16B DMA: LDS[l] = MEM[g]; tracked by ASYNCcnt, bypasses VGPRs
  __builtin_amdgcn_global_load_async_to_lds_b128(
      (AS1 v4i*)g, (AS3 v4i*)l, /*offset=*/0, /*cpol=*/0);
#else
  *(v8bf*)l = *(const v8bf*)g;
#endif
}

__device__ __forceinline__ void stage_fence() {
#ifdef USE_ASYNC_LDS
  __builtin_amdgcn_s_wait_asynccnt(0);
#endif
  __syncthreads();
}

// ---------------------------------------------------------------------------
// Weight convert + transpose: W[k][n] (f32, KxN row-major) -> Wt[n][k] (bf16)
// So GEMM B-fragments read 16 consecutive K values per lane (contiguous k order, same as the A frags).
// ---------------------------------------------------------------------------
__global__ __launch_bounds__(256) void wt_transpose_kernel(
    const float* __restrict__ W, __bf16* __restrict__ Wt, int K, int N) {
  __shared__ float tile[32][33];
  int kb = blockIdx.x * 32, nb = blockIdx.y * 32;
  int tx = threadIdx.x & 31, ty = threadIdx.x >> 5;  // 256 thr: ty in 0..7
#pragma unroll
  for (int i = 0; i < 32; i += 8)
    tile[ty + i][tx] = W[(size_t)(kb + ty + i) * N + (nb + tx)];
  __syncthreads();
#pragma unroll
  for (int i = 0; i < 32; i += 8)
    Wt[(size_t)(nb + ty + i) * K + (kb + tx)] = (__bf16)tile[tx][ty + i];
}

// ---------------------------------------------------------------------------
// LayerNorm over C=1024, one row per 256-thread block, bf16 output.
// ---------------------------------------------------------------------------
__global__ __launch_bounds__(256) void layernorm_bf16_kernel(
    const float* __restrict__ x, const float* __restrict__ g,
    const float* __restrict__ b, __bf16* __restrict__ out) {
  __shared__ float rs[256], rq[256];
  int row = blockIdx.x, tid = threadIdx.x;
  const float4 v = ((const float4*)(x + (size_t)row * Cdim))[tid];
  float s = v.x + v.y + v.z + v.w;
  float q = v.x * v.x + v.y * v.y + v.z * v.z + v.w * v.w;
  rs[tid] = s; rq[tid] = q;
  __syncthreads();
  for (int o = 128; o > 0; o >>= 1) {
    if (tid < o) { rs[tid] += rs[tid + o]; rq[tid] += rq[tid + o]; }
    __syncthreads();
  }
  float mean = rs[0] * (1.0f / Cdim);
  float var  = rq[0] * (1.0f / Cdim) - mean * mean;
  float inv  = rsqrtf(var + 1e-5f);
  float vv[4] = {v.x, v.y, v.z, v.w};
#pragma unroll
  for (int j = 0; j < 4; ++j) {
    int idx = tid * 4 + j;
    float val = (vv[j] - mean) * inv * g[idx] + b[idx];
    out[(size_t)row * Cdim + idx] = (__bf16)val;
  }
}

// ---------------------------------------------------------------------------
// Tiled WMMA GEMM: C[M,N] = A[M,K](bf16) * Bt[N,K](bf16)^T, f32 accumulate.
// Block tile 64(M) x 128(N), K-chunks of 32 through double-buffered LDS
// (async DMA to LDS when available). 8 waves: 4 M x 2 N; each wave owns a
// 16x64 output (4 accumulators).
// Epilogue: +bias, ReLU, +f32 residual, f32 and/or bf16 output,
// optional V-transpose layout [b][h][d][t] for attention.
// ---------------------------------------------------------------------------
#define GBM 64
#define GBN 128
#define GBK 32
#define KPAD 40

__global__ __launch_bounds__(256) void gemm_bf16_kernel(
    const __bf16* __restrict__ A, const __bf16* __restrict__ Bt,
    int M, int N, int K,
    const float* __restrict__ bias, const float* __restrict__ resid,
    int relu, float* __restrict__ outF, __bf16* __restrict__ outB, int vtrans) {
  __shared__ __bf16 sA[2][GBM][KPAD];
  __shared__ __bf16 sB[2][GBN][KPAD];
  const int tid = threadIdx.x, lane = tid & 31, wid = tid >> 5;
  const int wm = wid & 3, wn = wid >> 2;
  const int mBlock = blockIdx.y * GBM, nBlock = blockIdx.x * GBN;
  const int lm = lane & 15, lh = lane >> 4;

  v8f acc[4] = {};

  const int ar = tid >> 2, ac = (tid & 3) * 8;  // A stage: 64 rows x 4 chunks
  auto stage_tiles = [&](int buf, int kb) {
    stage16(&A[(size_t)(mBlock + ar) * K + kb + ac], &sA[buf][ar][ac]);
#pragma unroll
    for (int i = 0; i < 2; ++i) {  // B stage: 128 rows x 4 chunks, 2/thread
      int cid = tid * 2 + i;
      int r = cid >> 2, c = (cid & 3) * 8;
      stage16(&Bt[(size_t)(nBlock + r) * K + kb + c], &sB[buf][r][c]);
    }
#ifndef USE_ASYNC_LDS
    __builtin_prefetch(&A[(size_t)(mBlock + ar) * K + kb + GBK + ac], 0, 1);
    __builtin_prefetch(&Bt[(size_t)(nBlock + (tid >> 1)) * K + kb + GBK], 0, 1);
#endif
  };

  const int nk = K / GBK;
  stage_tiles(0, 0);
  stage_fence();

  for (int ik = 0; ik < nk; ++ik) {
    const int cur = ik & 1;
    if (ik + 1 < nk) stage_tiles(1 - cur, (ik + 1) * GBK);

    Frag16 fa;  // A frag: 16 consecutive K per lane, two b128 LDS loads
    const int am = wm * 16 + lm, ak = lh * 16;
    fa.h[0] = *(const v8bf*)&sA[cur][am][ak];
    fa.h[1] = *(const v8bf*)&sA[cur][am][ak + 8];
#pragma unroll
    for (int nt = 0; nt < 4; ++nt) {  // B frag: 16 consecutive K per lane
      Frag16 fb;
      const int bn = wn * 64 + nt * 16 + lm, bk = lh * 16;
      fb.h[0] = *(const v8bf*)&sB[cur][bn][bk];
      fb.h[1] = *(const v8bf*)&sB[cur][bn][bk + 8];
      acc[nt] = wmma_bf16(fa, fb, acc[nt]);
    }
    stage_fence();  // next buffer fully staged before it is consumed
  }

#pragma unroll
  for (int nt = 0; nt < 4; ++nt) {
#pragma unroll
    for (int r = 0; r < 8; ++r) {
      const int m = mBlock + wm * 16 + r + lh * 8;     // C/D layout rows
      const int n = nBlock + wn * 64 + nt * 16 + lm;   // C/D layout cols
      float v = acc[nt][r];
      if (bias)  v += bias[n];
      if (relu)  v = fmaxf(v, 0.0f);
      if (resid) v += resid[(size_t)m * N + n];
      if (outF)  outF[(size_t)m * N + n] = v;
      if (outB) {
        if (vtrans) {  // V -> [b][h][d][t] for attention P*V B-fragments
          int bb = m >> 10, t = m & (Tdim - 1);
          int hh = n >> 6, d = n & (Ddim - 1);
          outB[(((size_t)bb * Hdim + hh) * Ddim + d) * Tdim + t] = (__bf16)v;
        } else {
          outB[(size_t)m * N + n] = (__bf16)v;
        }
      }
    }
  }
}

// ---------------------------------------------------------------------------
// Flash-style attention. Block = 8 waves; each wave owns a 16-query tile.
// Q,K: bf16 [B*T][C] row-major (head slice contiguous in d).
// Vt:  bf16 [b][h][d][t]. Streams 32-key chunks with online softmax.
// ---------------------------------------------------------------------------
__global__ __launch_bounds__(256) void attn_kernel(
    const __bf16* __restrict__ Q, const __bf16* __restrict__ Kmat,
    const __bf16* __restrict__ Vt, const int* __restrict__ mask,
    __bf16* __restrict__ O) {
  __shared__ __bf16 sP[8][16][KPAD];  // per-wave P staging (C-layout -> A-layout)
  const int tid = threadIdx.x, lane = tid & 31, w = tid >> 5;
  const int lm = lane & 15, lh = lane >> 4;
  const int bh = blockIdx.x;
  const int b = bh >> 4, h = bh & (Hdim - 1);
  const int qbase = blockIdx.y * 128 + w * 16;
  const size_t headBase = (size_t)b * Tdim * Cdim + (size_t)h * Ddim;

  // Q fragments for the whole tile (A layout: m=lm, 16 consecutive K per lane)
  Frag16 qa0, qa1;
  {
    const __bf16* qr = Q + headBase + (size_t)(qbase + lm) * Cdim;
    const int ak = lh * 16;
    qa0.h[0] = *(const v8bf*)&qr[ak];
    qa0.h[1] = *(const v8bf*)&qr[ak + 8];
    qa1.h[0] = *(const v8bf*)&qr[32 + ak];
    qa1.h[1] = *(const v8bf*)&qr[32 + ak + 8];
  }

  v8f oc[4] = {};
  float mrun[8], lrun[8];
#pragma unroll
  for (int r = 0; r < 8; ++r) { mrun[r] = -1e30f; lrun[r] = 0.0f; }

  const __bf16* vrow = Vt + ((size_t)b * Hdim + h) * Ddim * Tdim;

  for (int kc = 0; kc < Tdim; kc += 32) {
    // ---- S = Q * K^T for 32 keys (two 16x16 tiles, K-dim = D = 64) ----
    v8f s0 = {}, s1 = {};
    {
      const __bf16* kr0 = Kmat + headBase + (size_t)(kc + lm) * Cdim;
      const __bf16* kr1 = Kmat + headBase + (size_t)(kc + 16 + lm) * Cdim;
      const int bk = lh * 16;
      Frag16 f;
      f.h[0] = *(const v8bf*)&kr0[bk];      f.h[1] = *(const v8bf*)&kr0[bk + 8];
      s0 = wmma_bf16(qa0, f, s0);
      f.h[0] = *(const v8bf*)&kr0[32 + bk]; f.h[1] = *(const v8bf*)&kr0[32 + bk + 8];
      s0 = wmma_bf16(qa1, f, s0);
      f.h[0] = *(const v8bf*)&kr1[bk];      f.h[1] = *(const v8bf*)&kr1[bk + 8];
      s1 = wmma_bf16(qa0, f, s1);
      f.h[0] = *(const v8bf*)&kr1[32 + bk]; f.h[1] = *(const v8bf*)&kr1[32 + bk + 8];
      s1 = wmma_bf16(qa1, f, s1);
    }

    // ---- scale, padding mask, online softmax ----
    const float msk0 = (mask[b * Tdim + kc + lm] == 0)      ? -1e30f : 0.0f;
    const float msk1 = (mask[b * Tdim + kc + 16 + lm] == 0) ? -1e30f : 0.0f;
    float corr[8];
#pragma unroll
    for (int r = 0; r < 8; ++r) {
      float a0 = s0[r] * 0.125f + msk0;  // 1/sqrt(64)
      float a1 = s1[r] * 0.125f + msk1;
      float mx = fmaxf(a0, a1);
      mx = fmaxf(mx, __shfl_xor(mx, 1));
      mx = fmaxf(mx, __shfl_xor(mx, 2));
      mx = fmaxf(mx, __shfl_xor(mx, 4));
      mx = fmaxf(mx, __shfl_xor(mx, 8));  // stays inside 16-lane half (row set)
      float mnew = fmaxf(mrun[r], mx);
      float cr = __expf(mrun[r] - mnew);
      float p0 = __expf(a0 - mnew);
      float p1 = __expf(a1 - mnew);
      float rs = p0 + p1;
      rs += __shfl_xor(rs, 1);
      rs += __shfl_xor(rs, 2);
      rs += __shfl_xor(rs, 4);
      rs += __shfl_xor(rs, 8);
      lrun[r] = lrun[r] * cr + rs;
      mrun[r] = mnew;
      corr[r] = cr;
      const int row = r + lh * 8;  // C/D layout row
      sP[w][row][lm]      = (__bf16)p0;
      sP[w][row][16 + lm] = (__bf16)p1;
    }
#pragma unroll
    for (int nt = 0; nt < 4; ++nt)
#pragma unroll
      for (int r = 0; r < 8; ++r) oc[nt][r] *= corr[r];
    __syncthreads();

    // ---- O += P * V  (P re-read in A layout, V B-frags contiguous in t) ----
    Frag16 pa;
    const int ak = lh * 16;
    pa.h[0] = *(const v8bf*)&sP[w][lm][ak];
    pa.h[1] = *(const v8bf*)&sP[w][lm][ak + 8];
    const int bk = lh * 16;
#pragma unroll
    for (int nt = 0; nt < 4; ++nt) {
      Frag16 fv;
      const __bf16* vr = vrow + (size_t)(nt * 16 + lm) * Tdim + kc;
      fv.h[0] = *(const v8bf*)&vr[bk];
      fv.h[1] = *(const v8bf*)&vr[bk + 8];
      oc[nt] = wmma_bf16(pa, fv, oc[nt]);
    }
    __syncthreads();
  }

  // ---- normalize and write O (row-major bf16 [B*T][C], head slice) ----
#pragma unroll
  for (int nt = 0; nt < 4; ++nt) {
#pragma unroll
    for (int r = 0; r < 8; ++r) {
      const int qrow = qbase + r + lh * 8;
      const int d = nt * 16 + lm;
      const float val = oc[nt][r] / lrun[r];
      O[headBase + (size_t)qrow * Cdim + d] = (__bf16)val;
    }
  }
}

// ---------------------------------------------------------------------------
// Host launcher
// ---------------------------------------------------------------------------
extern "C" void kernel_launch(void* const* d_in, const int* in_sizes, int n_in,
                              void* d_out, int out_size, void* d_ws, size_t ws_size,
                              hipStream_t stream) {
  (void)in_sizes; (void)n_in; (void)out_size; (void)ws_size;
  const float* x    = (const float*)d_in[0];
  const int*   mask = (const int*)d_in[1];
  const float* Wq   = (const float*)d_in[2];
  const float* Wk   = (const float*)d_in[3];
  const float* Wv   = (const float*)d_in[4];
  const float* Wo   = (const float*)d_in[5];
  const float* bo   = (const float*)d_in[6];
  const float* ln1g = (const float*)d_in[7];
  const float* ln1b = (const float*)d_in[8];
  const float* ln3g = (const float*)d_in[9];
  const float* ln3b = (const float*)d_in[10];
  const float* W1   = (const float*)d_in[11];
  const float* b1   = (const float*)d_in[12];
  const float* W2   = (const float*)d_in[13];
  const float* b2   = (const float*)d_in[14];
  float* out = (float*)d_out;

  char* ws = (char*)d_ws;
  size_t off = 0;
  auto alloc = [&](size_t bytes) -> char* {
    char* p = ws + off;
    off += (bytes + 255) & ~(size_t)255;
    return p;
  };

  __bf16* wq_t = (__bf16*)alloc((size_t)Cdim * Cdim * 2);
  __bf16* wk_t = (__bf16*)alloc((size_t)Cdim * Cdim * 2);
  __bf16* wv_t = (__bf16*)alloc((size_t)Cdim * Cdim * 2);
  __bf16* wo_t = (__bf16*)alloc((size_t)Cdim * Cdim * 2);
  __bf16* w1_t = (__bf16*)alloc((size_t)Fdim * Cdim * 2);  // [N=4C][K=C]
  __bf16* w2_t = (__bf16*)alloc((size_t)Cdim * Fdim * 2);  // [N=C][K=4C]
  __bf16* xn   = (__bf16*)alloc((size_t)Mrows * Cdim * 2);
  __bf16* qb   = (__bf16*)alloc((size_t)Mrows * Cdim * 2);
  __bf16* kb   = (__bf16*)alloc((size_t)Mrows * Cdim * 2);
  __bf16* vt   = (__bf16*)alloc((size_t)Mrows * Cdim * 2);  // [b][h][d][t]
  __bf16* ob   = (__bf16*)alloc((size_t)Mrows * Cdim * 2);
  float*  x1   = (float*) alloc((size_t)Mrows * Cdim * 4);
  __bf16* hn   = (__bf16*)alloc((size_t)Mrows * Cdim * 2);
  __bf16* h1   = (__bf16*)alloc((size_t)Mrows * Fdim * 2);

  const dim3 blk(256);
  // 1) weight convert + transpose to bf16
  wt_transpose_kernel<<<dim3(Cdim / 32, Cdim / 32), blk, 0, stream>>>(Wq, wq_t, Cdim, Cdim);
  wt_transpose_kernel<<<dim3(Cdim / 32, Cdim / 32), blk, 0, stream>>>(Wk, wk_t, Cdim, Cdim);
  wt_transpose_kernel<<<dim3(Cdim / 32, Cdim / 32), blk, 0, stream>>>(Wv, wv_t, Cdim, Cdim);
  wt_transpose_kernel<<<dim3(Cdim / 32, Cdim / 32), blk, 0, stream>>>(Wo, wo_t, Cdim, Cdim);
  wt_transpose_kernel<<<dim3(Cdim / 32, Fdim / 32), blk, 0, stream>>>(W1, w1_t, Cdim, Fdim);
  wt_transpose_kernel<<<dim3(Fdim / 32, Cdim / 32), blk, 0, stream>>>(W2, w2_t, Fdim, Cdim);

  // 2) pre-LN 1
  layernorm_bf16_kernel<<<Mrows, blk, 0, stream>>>(x, ln1g, ln1b, xn);

  // 3) Q,K,V projections (V written transposed per head)
  const dim3 gC(Cdim / GBN, Mrows / GBM);
  gemm_bf16_kernel<<<gC, blk, 0, stream>>>(xn, wq_t, Mrows, Cdim, Cdim,
                                           nullptr, nullptr, 0, nullptr, qb, 0);
  gemm_bf16_kernel<<<gC, blk, 0, stream>>>(xn, wk_t, Mrows, Cdim, Cdim,
                                           nullptr, nullptr, 0, nullptr, kb, 0);
  gemm_bf16_kernel<<<gC, blk, 0, stream>>>(xn, wv_t, Mrows, Cdim, Cdim,
                                           nullptr, nullptr, 0, nullptr, vt, 1);

  // 4) attention
  attn_kernel<<<dim3(Bdim * Hdim, Tdim / 128), blk, 0, stream>>>(qb, kb, vt, mask, ob);

  // 5) output projection + bias + residual -> x1 (f32)
  gemm_bf16_kernel<<<gC, blk, 0, stream>>>(ob, wo_t, Mrows, Cdim, Cdim,
                                           bo, x, 0, x1, nullptr, 0);

  // 6) pre-LN 2
  layernorm_bf16_kernel<<<Mrows, blk, 0, stream>>>(x1, ln3g, ln3b, hn);

  // 7) FFN
  gemm_bf16_kernel<<<dim3(Fdim / GBN, Mrows / GBM), blk, 0, stream>>>(
      hn, w1_t, Mrows, Fdim, Cdim, b1, nullptr, 1, nullptr, h1, 0);
  gemm_bf16_kernel<<<dim3(Cdim / GBN, Mrows / GBM), blk, 0, stream>>>(
      h1, w2_t, Mrows, Cdim, Fdim, b2, x1, 0, out, nullptr, 0);
}
